// AdaptiveECELoss_32195074850951
// MI455X (gfx1250) — matrix operand order repaired and checked by reading, not compiled, not run on hardware
//
#include <hip/hip_runtime.h>
#include <stdint.h>
#include <math.h>

// AdaptiveECELoss for MI455X (gfx1250, wave32).
// Pipeline:
//  1) rowstat: wave-per-row softmax-max (conf=1/sum(exp(l-max))), argmax==label.
//     Streams 512MB; async global->LDS double buffering (ASYNCcnt path).
//  2) exact quantile edges via 2-level radix-16 selection on float bit patterns
//     (positive floats are bit-order == numeric order). 256KB LDS histograms.
//  3) bin accumulation via V_WMMA_F32_16X16X32_F16: D += A(bin indicators) x
//     B(cols {1, acc, conf}); deterministic per-wave partials, serial combine.

typedef __attribute__((ext_vector_type(16))) _Float16 v16h;
typedef __attribute__((ext_vector_type(8)))  float    v8f;

#define ECE_NBINS 15

#if defined(__has_builtin)
#  if __has_builtin(__builtin_amdgcn_global_load_async_to_lds_b128) && \
      __has_builtin(__builtin_amdgcn_s_wait_asynccnt)
#    define ECE_USE_ASYNC 1
#  endif
#endif
#ifndef ECE_USE_ASYNC
#  define ECE_USE_ASYNC 0
#endif

// ---------------------------------------------------------------- utilities

__device__ __forceinline__ void wave_reduce_maxarg(float& m, int& mi) {
  #pragma unroll
  for (int off = 16; off > 0; off >>= 1) {
    float om = __shfl_xor(m, off, 32);
    int   oi = __shfl_xor(mi, off, 32);
    if (om > m || (om == m && oi < mi)) { m = om; mi = oi; }  // first occurrence
  }
}

__device__ __forceinline__ float wave_reduce_sum(float s) {
  #pragma unroll
  for (int off = 16; off > 0; off >>= 1) s += __shfl_xor(s, off, 32);
  return s;
}

#if ECE_USE_ASYNC
// builtin signature (from hipcc diagnostic): param0 = v4i (vector_size 16) ptr
// in global AS; param1 = LDS-side ptr; params 2,3 = constant offset / cpol.
typedef int v4i_vs __attribute__((vector_size(16)));
typedef __attribute__((address_space(1))) v4i_vs g_v4i;
typedef __attribute__((address_space(3))) v4i_vs l_v4i;

__device__ __forceinline__ void ece_async_row128(const float* gsrc, float* ldst, int lane) {
  const char* g = (const char*)gsrc + lane * 16;   // 32 lanes x 16B = 512B row
  char*       l = (char*)ldst + lane * 16;
  g_v4i* gp = (g_v4i*)(uintptr_t)g;
  l_v4i* lp = (l_v4i*)(unsigned)(uintptr_t)l;      // AS3 ptr = 32-bit LDS offset
  __builtin_amdgcn_global_load_async_to_lds_b128(gp, lp, 0, 0);
}
#endif

__device__ __forceinline__ void row_body_write(float4 v, int lane, int r,
                                               const int* __restrict__ labels,
                                               float* __restrict__ conf,
                                               float* __restrict__ accf) {
  float m = v.x; int mi = lane * 4;
  if (v.y > m) { m = v.y; mi = lane * 4 + 1; }
  if (v.z > m) { m = v.z; mi = lane * 4 + 2; }
  if (v.w > m) { m = v.w; mi = lane * 4 + 3; }
  wave_reduce_maxarg(m, mi);
  float s = __expf(v.x - m) + __expf(v.y - m) + __expf(v.z - m) + __expf(v.w - m);
  s = wave_reduce_sum(s);
  if (lane == 0) {
    conf[r] = 1.0f / s;                       // max prob = exp(0)/sum
    accf[r] = (mi == labels[r]) ? 1.0f : 0.0f;
  }
}

// ------------------------------------------------- kernel 1: per-row stats

__global__ void ece_rowstat128(const float* __restrict__ logits,
                               const int* __restrict__ labels,
                               float* __restrict__ conf, float* __restrict__ accf,
                               int N) {
  const int lane = threadIdx.x & 31;
  // wave-uniform row index, forced scalar so the row loop is a scalar branch
  const int w  = __builtin_amdgcn_readfirstlane(
                   (int)((blockIdx.x * blockDim.x + threadIdx.x) >> 5));
  const int nw = (int)((gridDim.x * blockDim.x) >> 5);
#if ECE_USE_ASYNC
  __shared__ __align__(16) float lbuf[8][2][128];   // 8 waves/block, double buffer
  const int wib = threadIdx.x >> 5;
  float* buf0 = &lbuf[wib][0][0];
  float* buf1 = &lbuf[wib][1][0];
  int r = w, p = 0;
  if (r < N) ece_async_row128(logits + (size_t)r * 128, buf0, lane);
  for (; r < N; r += nw) {
    int rn = r + nw;
    if (rn < N) {
      ece_async_row128(logits + (size_t)rn * 128, p ? buf0 : buf1, lane);
      __builtin_amdgcn_s_wait_asynccnt(1);   // current row's copy complete
    } else {
      __builtin_amdgcn_s_wait_asynccnt(0);
    }
    asm volatile("" ::: "memory");
    float4 v = ((const float4*)(p ? buf1 : buf0))[lane];
    row_body_write(v, lane, r, labels, conf, accf);
    p ^= 1;
  }
#else
  for (int r = w; r < N; r += nw) {
    const float* row = logits + (size_t)r * 128;
    __builtin_prefetch(row + (size_t)nw * 128 + lane * 4, 0, 3);
    float4 v = ((const float4*)row)[lane];
    row_body_write(v, lane, r, labels, conf, accf);
  }
#endif
}

// generic C (not used for the reference's C=128, kept for safety)
__global__ void ece_rowstat_generic(const float* __restrict__ logits,
                                    const int* __restrict__ labels,
                                    float* __restrict__ conf, float* __restrict__ accf,
                                    int N, int C) {
  const int lane = threadIdx.x & 31;
  const int w  = __builtin_amdgcn_readfirstlane(
                   (int)((blockIdx.x * blockDim.x + threadIdx.x) >> 5));
  const int nw = (int)((gridDim.x * blockDim.x) >> 5);
  for (int r = w; r < N; r += nw) {
    const float* row = logits + (size_t)r * C;
    float m = -INFINITY; int mi = 0x7fffffff;
    for (int j = lane; j < C; j += 32) { float x = row[j]; if (x > m) { m = x; mi = j; } }
    wave_reduce_maxarg(m, mi);
    float s = 0.0f;
    for (int j = lane; j < C; j += 32) s += __expf(row[j] - m);
    s = wave_reduce_sum(s);
    if (lane == 0) { conf[r] = 1.0f / s; accf[r] = (mi == labels[r]) ? 1.0f : 0.0f; }
  }
}

// ------------------------------------------ kernels 2..5: exact quantiles

__global__ void ece_zero_u32(unsigned* p, int n) {
  int i = blockIdx.x * blockDim.x + threadIdx.x;
  if (i < n) p[i] = 0u;
}

__global__ void ece_hist_hi(const float* __restrict__ conf, unsigned* __restrict__ hist, int N) {
  int i = blockIdx.x * blockDim.x + threadIdx.x;
  int st = gridDim.x * blockDim.x;
  for (; i < N; i += st) {
    unsigned b = __float_as_uint(conf[i]) >> 16;
    atomicAdd(&hist[b], 1u);
  }
}

// one workgroup: CDF of 65536-bin histogram in LDS, then find bucket+local rank
// for the 32 order statistics needed by the 16 interp'd edges.
__global__ void ece_cdf_rank(const unsigned* __restrict__ hist,
                             unsigned* __restrict__ buckets, unsigned* __restrict__ lranks,
                             int N) {
  extern __shared__ unsigned cdf[];        // 65536 u32 = 256KB (320KB LDS/WGP)
  __shared__ unsigned chunkv[1024];
  const int t = threadIdx.x;
  unsigned run = 0;
  for (int j = 0; j < 64; ++j) { run += hist[t * 64 + j]; cdf[t * 64 + j] = run; }
  chunkv[t] = run;
  __syncthreads();
  if (t == 0) {
    unsigned c = 0;
    for (int k = 0; k < 1024; ++k) { unsigned tmp = chunkv[k]; chunkv[k] = c; c += tmp; }
  }
  __syncthreads();
  unsigned add = chunkv[t];
  for (int j = 0; j < 64; ++j) cdf[t * 64 + j] += add;
  __syncthreads();
  if (t < 32) {
    int k = t >> 1;
    float step = (float)N / (float)ECE_NBINS;
    float q = step * (float)k;
    unsigned rank;
    if (q >= (float)(N - 1)) rank = (unsigned)(N - 1);
    else {
      unsigned i = (unsigned)q;
      rank = i + (unsigned)(t & 1);
      if (rank > (unsigned)(N - 1)) rank = (unsigned)(N - 1);
    }
    unsigned lo = 0, hi = 65535;
    while (lo < hi) { unsigned mid = (lo + hi) >> 1; if (cdf[mid] > rank) hi = mid; else lo = mid + 1; }
    buckets[t] = lo;
    lranks[t]  = rank - (lo ? cdf[lo - 1] : 0u);
  }
}

// one workgroup per rank: low-16-bit histogram restricted to the rank's high
// bucket fully resolves the exact float bit pattern of that order statistic.
__global__ void ece_select_low(const float* __restrict__ conf,
                               const unsigned* __restrict__ buckets,
                               const unsigned* __restrict__ lranks,
                               float* __restrict__ selval, int N) {
  extern __shared__ unsigned lh[];         // 65536 u32 in LDS
  __shared__ unsigned psum[1024];
  const int t = threadIdx.x;
  const unsigned b  = buckets[blockIdx.x];
  const unsigned lr = lranks[blockIdx.x];
  for (int j = t; j < 65536; j += 1024) lh[j] = 0u;
  __syncthreads();
  for (int i = t; i < N; i += 1024) {
    unsigned bits = __float_as_uint(conf[i]);
    if ((bits >> 16) == b) atomicAdd(&lh[bits & 0xFFFFu], 1u);
  }
  __syncthreads();
  unsigned s = 0;
  for (int j = 0; j < 64; ++j) s += lh[t * 64 + j];
  psum[t] = s;
  __syncthreads();
  if (t == 0) {
    unsigned cum = 0; int chunk = 0;
    for (; chunk < 1024; ++chunk) { if (cum + psum[chunk] > lr) break; cum += psum[chunk]; }
    unsigned low = 0;
    for (int j = 0; j < 64; ++j) {
      unsigned h = lh[chunk * 64 + j];
      if (cum + h > lr) { low = (unsigned)(chunk * 64 + j); break; }
      cum += h;
    }
    selval[blockIdx.x] = __uint_as_float(((unsigned)b << 16) | low);
  }
}

__global__ void ece_edges(const float* __restrict__ selval, float* __restrict__ edges, int N) {
  int k = threadIdx.x;
  if (k < ECE_NBINS + 1) {
    float step = (float)N / (float)ECE_NBINS;
    float q = step * (float)k;
    float e;
    if (q >= (float)(N - 1)) e = selval[2 * k];
    else {
      float fl = floorf(q), fr = q - fl;
      e = selval[2 * k] * (1.0f - fr) + selval[2 * k + 1] * fr;
    }
    edges[k] = e;
  }
}

// --------------------------------- kernel 6: WMMA bin accumulation (f16 A/B)
// D(16x16,f32) += A(16x32) x B(32x16); A[m][k]=1 if elem k lands in bin m,
// B[k][0]=1 (count), B[k][1]=acc_k, B[k][2]=conf_k. 32 elems per WMMA.

__global__ void ece_bin_accum(const float* __restrict__ conf, const float* __restrict__ accf,
                              const float* __restrict__ edges, float* __restrict__ partial,
                              int N) {
  const int lane = threadIdx.x & 31;
  // wave-uniform chunk index as a scalar -> uniform loop, EXEC all-1s at WMMA
  const int wid = __builtin_amdgcn_readfirstlane(
                    (int)((blockIdx.x * blockDim.x + threadIdx.x) >> 5));
  const int nw  = (int)((gridDim.x * blockDim.x) >> 5);
  const int m = lane & 15;          // row (bin) for A, column for B
  const int h = lane >> 4;
  // row 15 unused: +inf lower edge makes its indicators identically 0
  const float eL = (m < ECE_NBINS) ? edges[m] : INFINITY;
  const float eU = (m < ECE_NBINS) ? edges[m + 1] : 0.0f;
  // branch-free per-lane column masks for B: col0 = count, col1 = acc, col2 = conf
  const float s0 = (m == 0) ? 1.0f : 0.0f;
  const float s1 = (m == 1) ? 1.0f : 0.0f;
  const float s2 = (m == 2) ? 1.0f : 0.0f;
  v8f c = {};
  const int step = nw * 32;
  for (int base = wid * 32; base < N; base += step) {
    int i = base + lane;
    int idx = (i < N) ? i : (N - 1);        // clamped, always-executed loads
    float cl = conf[idx];
    float al = accf[idx];
    float cv = (i < N) ? cl : -1.0f;        // -1 never lands in a bin (edges > 0)
    float av = (i < N) ? al : 0.0f;
    v16h A, B;
    #pragma unroll
    for (int t = 0; t < 16; ++t) {
      // A-matrix 16x32 f16 layout: lane(l&15)=M; K = (t&7) + 8*(l>>4) + (t>>3)*16
      int Ka = (t & 7) + 8 * h + ((t >> 3) << 4);
      float ck = __shfl(cv, Ka, 32);
      A[t] = ((ck > eL) && (ck <= eU)) ? (_Float16)1.0f : (_Float16)0.0f;
      // B-matrix 32x16 f16 layout: lane(l&15)=N col; K = (t&7) + 8*(l>>4) + (t>>3)*16
      int Kb = (t & 7) + 8 * h + ((t >> 3) << 4);
      float ckb = __shfl(cv, Kb, 32);
      float akb = __shfl(av, Kb, 32);
      B[t] = (_Float16)fmaf(s1, akb, fmaf(s2, ckb, s0));
    }
    c = __builtin_amdgcn_wmma_f32_16x16x32_f16(false, A, false, B, (short)0, c, false, false);
  }
  // D layout: VGPR r -> lanes0-15: M=r,N=lane; lanes16-31: M=8+r,N=lane-16
  float* pp = partial + (size_t)wid * 48;
  for (int mm = 0; mm < ECE_NBINS; ++mm) {
    int r  = (mm < 8) ? mm : (mm - 8);
    int lb = (mm < 8) ? 0 : 16;
    float cnt = __shfl(c[r], lb + 0, 32);
    float sa  = __shfl(c[r], lb + 1, 32);
    float sc  = __shfl(c[r], lb + 2, 32);
    if (lane == 0) { pp[mm * 3 + 0] = cnt; pp[mm * 3 + 1] = sa; pp[mm * 3 + 2] = sc; }
  }
}

__global__ void ece_finalize(const float* __restrict__ partial, int nwaves,
                             float* __restrict__ out, int N) {
  __shared__ float sums[48];
  int t = threadIdx.x;
  if (t < 45) {                      // deterministic serial combine
    float s = 0.0f;
    for (int w = 0; w < nwaves; ++w) s += partial[(size_t)w * 48 + t];
    sums[t] = s;
  }
  __syncthreads();
  if (t == 0) {
    float ece = 0.0f, n = (float)N;
    for (int mm = 0; mm < ECE_NBINS; ++mm) {
      float cnt = sums[mm * 3 + 0], sa = sums[mm * 3 + 1], sc = sums[mm * 3 + 2];
      if (cnt > 0.0f) {
        float denom = fmaxf(cnt, 1.0f);
        ece += fabsf(sc / denom - sa / denom) * (cnt / n);
      }
    }
    out[0] = ece;
  }
}

// ---------------------------------------------------------------- launcher

extern "C" void kernel_launch(void* const* d_in, const int* in_sizes, int n_in,
                              void* d_out, int out_size, void* d_ws, size_t ws_size,
                              hipStream_t stream) {
  (void)n_in; (void)out_size; (void)ws_size;
  const float* logits = (const float*)d_in[0];
  const int*   labels = (const int*)d_in[1];
  const int N = in_sizes[1];
  const int C = in_sizes[0] / (N > 0 ? N : 1);
  float* out = (float*)d_out;

  char* ws = (char*)d_ws;
  auto a256 = [](size_t x) { return (x + 255) & ~(size_t)255; };
  size_t o = 0;
  float*    conf    = (float*)(ws + o);    o = a256(o + (size_t)4 * N);
  float*    accf    = (float*)(ws + o);    o = a256(o + (size_t)4 * N);
  unsigned* hist    = (unsigned*)(ws + o); o = a256(o + 65536u * 4u);
  unsigned* buckets = (unsigned*)(ws + o); o = a256(o + 32u * 4u);
  unsigned* lranks  = (unsigned*)(ws + o); o = a256(o + 32u * 4u);
  float*    selval  = (float*)(ws + o);    o = a256(o + 32u * 4u);
  float*    edges   = (float*)(ws + o);    o = a256(o + 16u * 4u);
  const int ACC_BLOCKS = 512, ACC_TPB = 256;
  const int ACC_WAVES  = ACC_BLOCKS * ACC_TPB / 32;
  float*    partial = (float*)(ws + o);    o += (size_t)ACC_WAVES * 48u * 4u;

  ece_zero_u32<<<64, 1024, 0, stream>>>(hist, 65536);

  if (C == 128)
    ece_rowstat128<<<4096, 256, 0, stream>>>(logits, labels, conf, accf, N);
  else
    ece_rowstat_generic<<<4096, 256, 0, stream>>>(logits, labels, conf, accf, N, C);

  ece_hist_hi<<<1024, 256, 0, stream>>>(conf, hist, N);
  ece_cdf_rank<<<1, 1024, 65536 * sizeof(unsigned), stream>>>(hist, buckets, lranks, N);
  ece_select_low<<<32, 1024, 65536 * sizeof(unsigned), stream>>>(conf, buckets, lranks, selval, N);
  ece_edges<<<1, 32, 0, stream>>>(selval, edges, N);
  ece_bin_accum<<<ACC_BLOCKS, ACC_TPB, 0, stream>>>(conf, accf, edges, partial, N);
  ece_finalize<<<1, 64, 0, stream>>>(partial, ACC_WAVES, out, N);
}
